// Omniformer_69956427317725
// MI455X (gfx1250) — compile-verified, matrix operand order checked by reading, not run on hardware
//
#include <hip/hip_runtime.h>

typedef __attribute__((ext_vector_type(16))) __bf16 v16bf;
typedef __attribute__((ext_vector_type(8)))  float  v8f;

#define B_     32
#define S_     512
#define INDIM  256
#define CTX_   128
#define D_     128
#define DH_    32
#define HID_   512
#define TOTAL_ 181248

// offsets into per-batch flat vector (element units)
#define OFF_AW  0
#define OFF_AB  49152
#define OFF_F1W 49536
#define OFF_F1B 115072
#define OFF_F2W 115584
#define OFF_F2B 181120

__device__ __forceinline__ __bf16 f2bf(float f) { return (__bf16)f; }

// One WMMA K-step (K += 32) for C(16x16) += A(16x32) * B(32x16).
// arow: pointer to this lane's A row (row index = lane&15), contiguous K.
// bcol: pointer to this lane's B column (col index = lane&15), contiguous K
//       (i.e. B stored N-major: element (k,n) at bcol_n[k]).
__device__ __forceinline__ void wmma_step(const float* __restrict__ arow,
                                          const float* __restrict__ bcol,
                                          int kk, int half, v8f& acc) {
  v16bf a, b;
#pragma unroll
  for (int j = 0; j < 8; ++j) {
    a[j]     = f2bf(arow[kk + half * 8 + j]);
    a[j + 8] = f2bf(arow[kk + 16 + half * 8 + j]);
    b[j]     = f2bf(bcol[kk + half * 8 + j]);
    b[j + 8] = f2bf(bcol[kk + 16 + half * 8 + j]);
  }
  acc = __builtin_amdgcn_wmma_f32_16x16x32_bf16(false, a, false, b,
                                                (short)0, acc, false, false);
}

// ---------------- h = x @ in_proj_w^T + b + pe ----------------
// grid: 1024 blocks x 256 thr (8 waves); tiles: 1024 m-tiles x 8 n-tiles
__global__ void k_in_proj(const float* __restrict__ x, const float* __restrict__ w,
                          const float* __restrict__ bias, const float* __restrict__ pe,
                          float* __restrict__ h) {
  int wave = threadIdx.x >> 5, lane = threadIdx.x & 31;
  int tile = blockIdx.x * 8 + wave;
  int nt = tile & 7, mt = tile >> 3;
  int half = lane >> 4, idx = lane & 15;
  const float* arow = x + (size_t)(mt * 16 + idx) * INDIM;
  const float* bcol = w + (size_t)(nt * 16 + idx) * INDIM;
  v8f acc = {};
  for (int kk = 0; kk < INDIM; kk += 32) wmma_step(arow, bcol, kk, half, acc);
  int col = nt * 16 + idx;
  float bv = bias[col];
#pragma unroll
  for (int r = 0; r < 8; ++r) {
    int rowg = mt * 16 + half * 8 + r;
    int s = rowg & (S_ - 1);
    h[(size_t)rowg * D_ + col] = acc[r] + bv + pe[(size_t)s * D_ + col];
  }
}

// ---------------- hid = relu(ctx @ hfc1_w[l]^T + b) ----------------
// grid: 8 blocks x 256 thr; tiles: 2 m x 32 n
__global__ void k_fc1(const float* __restrict__ ctxv, const float* __restrict__ w,
                      const float* __restrict__ bias, float* __restrict__ hid, int l) {
  int wave = threadIdx.x >> 5, lane = threadIdx.x & 31;
  int tile = blockIdx.x * 8 + wave;
  int nt = tile & 31, mt = tile >> 5;
  int half = lane >> 4, idx = lane & 15;
  const float* arow = ctxv + (size_t)(mt * 16 + idx) * CTX_;
  const float* bcol = w + (size_t)l * HID_ * CTX_ + (size_t)(nt * 16 + idx) * CTX_;
  v8f acc = {};
  for (int kk = 0; kk < CTX_; kk += 32) wmma_step(arow, bcol, kk, half, acc);
  int col = nt * 16 + idx;
  float bv = bias[l * HID_ + col];
#pragma unroll
  for (int r = 0; r < 8; ++r)
    hid[(size_t)(mt * 16 + half * 8 + r) * HID_ + col] = fmaxf(acc[r] + bv, 0.f);
}

// ---------------- flat = hid @ hfc2_w[l]^T + b  (the big 742MB stream) -------
// grid: 1416 blocks x 256 thr; one wave per n-tile, computes BOTH m-tiles
// so each weight element is read exactly once.
__global__ void k_fc2(const float* __restrict__ hid, const float* __restrict__ w,
                      const float* __restrict__ bias, float* __restrict__ flat, int l) {
  int wave = threadIdx.x >> 5, lane = threadIdx.x & 31;
  int nt = blockIdx.x * 8 + wave;            // 11328 n-tiles
  int half = lane >> 4, idx = lane & 15;
  const float* a0 = hid + (size_t)idx * HID_;
  const float* a1 = hid + (size_t)(16 + idx) * HID_;
  const float* bcol = w + (size_t)l * TOTAL_ * HID_ + (size_t)(nt * 16 + idx) * HID_;
  v8f acc0 = {}, acc1 = {};
  for (int kk = 0; kk < HID_; kk += 32) {
    v16bf av0, av1, bv;
#pragma unroll
    for (int j = 0; j < 8; ++j) {
      bv[j]      = f2bf(bcol[kk + half * 8 + j]);
      bv[j + 8]  = f2bf(bcol[kk + 16 + half * 8 + j]);
      av0[j]     = f2bf(a0[kk + half * 8 + j]);
      av0[j + 8] = f2bf(a0[kk + 16 + half * 8 + j]);
      av1[j]     = f2bf(a1[kk + half * 8 + j]);
      av1[j + 8] = f2bf(a1[kk + 16 + half * 8 + j]);
    }
    acc0 = __builtin_amdgcn_wmma_f32_16x16x32_bf16(false, av0, false, bv, (short)0, acc0, false, false);
    acc1 = __builtin_amdgcn_wmma_f32_16x16x32_bf16(false, av1, false, bv, (short)0, acc1, false, false);
  }
  int col = nt * 16 + idx;
  float bv = bias[(size_t)l * TOTAL_ + col];
#pragma unroll
  for (int r = 0; r < 8; ++r) {
    int m = half * 8 + r;
    flat[(size_t)m * TOTAL_ + col]        = acc0[r] + bv;
    flat[(size_t)(16 + m) * TOTAL_ + col] = acc1[r] + bv;
  }
}

// ---------------- qkv = h @ aw[b]^T + ab[b] (per-batch hyper-weights) --------
// grid: 3072 blocks x 256 thr; tiles: 24 n x 32 m x 32 b
__global__ void k_qkv(const float* __restrict__ h, const float* __restrict__ flat,
                      float* __restrict__ qkv) {
  int wave = threadIdx.x >> 5, lane = threadIdx.x & 31;
  int tile = blockIdx.x * 8 + wave;
  int nt = tile % 24;
  int mt = (tile / 24) & 31;
  int b  = tile / (24 * 32);
  int half = lane >> 4, idx = lane & 15;
  const float* fb = flat + (size_t)b * TOTAL_;
  const float* arow = h + ((size_t)b * S_ + mt * 16 + idx) * D_;
  const float* bcol = fb + OFF_AW + (size_t)(nt * 16 + idx) * D_;
  v8f acc = {};
  for (int kk = 0; kk < D_; kk += 32) wmma_step(arow, bcol, kk, half, acc);
  int col = nt * 16 + idx;
  float bv = fb[OFF_AB + col];
#pragma unroll
  for (int r = 0; r < 8; ++r)
    qkv[((size_t)b * S_ + mt * 16 + half * 8 + r) * (3 * D_) + col] = acc[r] + bv;
}

// ---------------- attention: one wave per (b, head, 16-row q-tile) ----------
// grid: 4096 blocks x 32 thr; scores tile in LDS, in-wave softmax, WMMA P@V
__global__ void k_attn(const float* __restrict__ qkv, float* __restrict__ ctxout) {
  __shared__ float sc[16][S_ + 8];
  int lane = threadIdx.x;
  int bid = blockIdx.x;
  int qt = bid & 31;
  int hh = (bid >> 5) & 3;
  int b  = bid >> 7;
  int half = lane >> 4, idx = lane & 15;
  const float* qb = qkv + (size_t)b * S_ * (3 * D_);
  const float scale = 0.17677669529663689f;  // 1/sqrt(DH=32)

  // scores(16 x 512) = Q(16x32) @ K^T(32x512), K-dim = DH = one WMMA step
  const float* arow = qb + (size_t)(qt * 16 + idx) * (3 * D_) + hh * DH_;
  for (int nt = 0; nt < 32; ++nt) {
    const float* bcol = qb + (size_t)(nt * 16 + idx) * (3 * D_) + D_ + hh * DH_;
    v8f acc = {};
    wmma_step(arow, bcol, 0, half, acc);
#pragma unroll
    for (int r = 0; r < 8; ++r)
      sc[half * 8 + r][nt * 16 + idx] = acc[r] * scale;
  }
  __syncthreads();

  // softmax per row (lanes 0..15 each own one row)
  if (lane < 16) {
    float m = -1e30f;
    for (int j = 0; j < S_; ++j) m = fmaxf(m, sc[lane][j]);
    float s = 0.f;
    for (int j = 0; j < S_; ++j) { float e = __expf(sc[lane][j] - m); sc[lane][j] = e; s += e; }
    float inv = 1.f / s;
    for (int j = 0; j < S_; ++j) sc[lane][j] *= inv;
  }
  __syncthreads();

  // ctx(16 x 32) = P(16x512) @ V(512x32); V is K-major in qkv
  const float* vb = qb + 2 * D_ + hh * DH_;
  for (int nt = 0; nt < 2; ++nt) {
    v8f acc = {};
    for (int kk = 0; kk < S_; kk += 32) {
      v16bf a, bv;
#pragma unroll
      for (int j = 0; j < 8; ++j) {
        a[j]      = f2bf(sc[idx][kk + half * 8 + j]);
        a[j + 8]  = f2bf(sc[idx][kk + 16 + half * 8 + j]);
        bv[j]     = f2bf(vb[(size_t)(kk + half * 8 + j) * (3 * D_) + nt * 16 + idx]);
        bv[j + 8] = f2bf(vb[(size_t)(kk + 16 + half * 8 + j) * (3 * D_) + nt * 16 + idx]);
      }
      acc = __builtin_amdgcn_wmma_f32_16x16x32_bf16(false, a, false, bv, (short)0, acc, false, false);
    }
#pragma unroll
    for (int r = 0; r < 8; ++r)
      ctxout[((size_t)b * S_ + qt * 16 + half * 8 + r) * D_ + hh * DH_ + nt * 16 + idx] = acc[r];
  }
}

// ---------------- y = h + alpha1[l] * (ctx @ out_proj_w[l]^T + b) -----------
// grid: 1024 blocks x 256 thr
__global__ void k_outproj(const float* __restrict__ ctxin, const float* __restrict__ w,
                          const float* __restrict__ bias, const float* __restrict__ alpha,
                          const float* __restrict__ h, float* __restrict__ y, int l) {
  int wave = threadIdx.x >> 5, lane = threadIdx.x & 31;
  int tile = blockIdx.x * 8 + wave;
  int nt = tile & 7, mt = tile >> 3;
  int half = lane >> 4, idx = lane & 15;
  const float* arow = ctxin + (size_t)(mt * 16 + idx) * D_;
  const float* bcol = w + (size_t)l * D_ * D_ + (size_t)(nt * 16 + idx) * D_;
  v8f acc = {};
  for (int kk = 0; kk < D_; kk += 32) wmma_step(arow, bcol, kk, half, acc);
  int col = nt * 16 + idx;
  float bv = bias[l * D_ + col];
  float al = alpha[l];
#pragma unroll
  for (int r = 0; r < 8; ++r) {
    size_t o = (size_t)(mt * 16 + half * 8 + r) * D_ + col;
    y[o] = h[o] + al * (acc[r] + bv);
  }
}

// ---------------- layernorm over D=128, one thread per row ------------------
__global__ void k_ln(const float* __restrict__ y, const float* __restrict__ g,
                     const float* __restrict__ bb, float* __restrict__ out, int l) {
  int row = blockIdx.x * blockDim.x + threadIdx.x;
  if (row >= B_ * S_) return;
  const float* p = y + (size_t)row * D_;
  float mu = 0.f;
  for (int j = 0; j < D_; ++j) mu += p[j];
  mu *= (1.f / D_);
  float var = 0.f;
  for (int j = 0; j < D_; ++j) { float d = p[j] - mu; var += d * d; }
  var *= (1.f / D_);
  float rs = rsqrtf(var + 1e-5f);
  float* o = out + (size_t)row * D_;
  for (int j = 0; j < D_; ++j)
    o[j] = (p[j] - mu) * rs * g[l * D_ + j] + bb[l * D_ + j];
}

// ---------------- ff = relu(h @ f1w[b]^T + f1b[b]) --------------------------
// grid: 4096 blocks x 256 thr; tiles: 32 n x 32 m x 32 b
__global__ void k_ffn1(const float* __restrict__ h, const float* __restrict__ flat,
                       float* __restrict__ ff) {
  int wave = threadIdx.x >> 5, lane = threadIdx.x & 31;
  int tile = blockIdx.x * 8 + wave;
  int nt = tile & 31, mt = (tile >> 5) & 31, b = tile >> 10;
  int half = lane >> 4, idx = lane & 15;
  const float* fb = flat + (size_t)b * TOTAL_;
  const float* arow = h + ((size_t)b * S_ + mt * 16 + idx) * D_;
  const float* bcol = fb + OFF_F1W + (size_t)(nt * 16 + idx) * D_;
  v8f acc = {};
  for (int kk = 0; kk < D_; kk += 32) wmma_step(arow, bcol, kk, half, acc);
  int col = nt * 16 + idx;
  float bv = fb[OFF_F1B + col];
#pragma unroll
  for (int r = 0; r < 8; ++r)
    ff[((size_t)b * S_ + mt * 16 + half * 8 + r) * (4 * D_) + col] = fmaxf(acc[r] + bv, 0.f);
}

// ---------------- y = h + alpha2[l] * (ff @ f2w[b]^T + f2b[b]) --------------
// grid: 1024 blocks x 256 thr; tiles: 8 n x 32 m x 32 b
__global__ void k_ffn2(const float* __restrict__ ff, const float* __restrict__ flat,
                       const float* __restrict__ alpha, const float* __restrict__ h,
                       float* __restrict__ y, int l) {
  int wave = threadIdx.x >> 5, lane = threadIdx.x & 31;
  int tile = blockIdx.x * 8 + wave;
  int nt = tile & 7, mt = (tile >> 3) & 31, b = tile >> 8;
  int half = lane >> 4, idx = lane & 15;
  const float* fb = flat + (size_t)b * TOTAL_;
  const float* arow = ff + ((size_t)b * S_ + mt * 16 + idx) * (4 * D_);
  const float* bcol = fb + OFF_F2W + (size_t)(nt * 16 + idx) * (4 * D_);
  v8f acc = {};
  for (int kk = 0; kk < 4 * D_; kk += 32) wmma_step(arow, bcol, kk, half, acc);
  int col = nt * 16 + idx;
  float bv = fb[OFF_F2B + col];
  float al = alpha[l];
#pragma unroll
  for (int r = 0; r < 8; ++r) {
    size_t o = ((size_t)b * S_ + mt * 16 + half * 8 + r) * D_ + col;
    y[o] = h[o] + al * (acc[r] + bv);
  }
}

// ---------------- out[b] = h[b, S-1, :] . head_w + head_b -------------------
__global__ void k_head(const float* __restrict__ h, const float* __restrict__ hw,
                       const float* __restrict__ hb, float* __restrict__ out) {
  int b = threadIdx.x;
  if (b < B_) {
    const float* p = h + ((size_t)b * S_ + (S_ - 1)) * D_;
    float s = 0.f;
    for (int j = 0; j < D_; ++j) s += p[j] * hw[j];
    out[b] = s + hb[0];
  }
}

extern "C" void kernel_launch(void* const* d_in, const int* in_sizes, int n_in,
                              void* d_out, int out_size, void* d_ws, size_t ws_size,
                              hipStream_t stream) {
  (void)in_sizes; (void)n_in; (void)out_size; (void)ws_size;
  const float* x    = (const float*)d_in[0];
  const float* ctxv = (const float*)d_in[1];
  const float* ipw  = (const float*)d_in[2];
  const float* ipb  = (const float*)d_in[3];
  const float* pe   = (const float*)d_in[4];
  const float* h1w  = (const float*)d_in[5];
  const float* h1b  = (const float*)d_in[6];
  const float* h2w  = (const float*)d_in[7];
  const float* h2b  = (const float*)d_in[8];
  const float* opw  = (const float*)d_in[9];
  const float* opb  = (const float*)d_in[10];
  const float* l1g  = (const float*)d_in[11];
  const float* l1b  = (const float*)d_in[12];
  const float* l2g  = (const float*)d_in[13];
  const float* l2b  = (const float*)d_in[14];
  const float* a1   = (const float*)d_in[15];
  const float* a2   = (const float*)d_in[16];
  const float* hw   = (const float*)d_in[17];
  const float* hb   = (const float*)d_in[18];

  float* ws = (float*)d_ws;
  size_t off = 0;
  float* h    = ws + off; off += (size_t)B_ * S_ * D_;       // 2M floats
  float* y    = ws + off; off += (size_t)B_ * S_ * D_;       // 2M
  float* qkv  = ws + off; off += (size_t)B_ * S_ * 3 * D_;   // 6M
  float* ctxo = ws + off; off += (size_t)B_ * S_ * D_;       // 2M
  float* ffb  = ws + off; off += (size_t)B_ * S_ * 4 * D_;   // 8M
  float* hid  = ws + off; off += (size_t)B_ * HID_;          // 16K
  float* flat = ws + off;                                    // B*TOTAL ~5.8M

  k_in_proj<<<1024, 256, 0, stream>>>(x, ipw, ipb, pe, h);
  for (int l = 0; l < 2; ++l) {
    k_fc1<<<8, 256, 0, stream>>>(ctxv, h1w, h1b, hid, l);
    k_fc2<<<1416, 256, 0, stream>>>(hid, h2w, h2b, flat, l);
    k_qkv<<<3072, 256, 0, stream>>>(h, flat, qkv);
    k_attn<<<4096, 32, 0, stream>>>(qkv, ctxo);
    k_outproj<<<1024, 256, 0, stream>>>(ctxo, opw, opb, a1, h, y, l);
    k_ln<<<64, 256, 0, stream>>>(y, l1g, l1b, h, l);
    k_ffn1<<<4096, 256, 0, stream>>>(h, flat, ffb);
    k_ffn2<<<1024, 256, 0, stream>>>(ffb, flat, a2, h, y, l);
    k_ln<<<64, 256, 0, stream>>>(y, l2g, l2b, h, l);
  }
  k_head<<<1, 32, 0, stream>>>(h, hw, hb, (float*)d_out);
}
